// MultiHeadAttention_83399674954286
// MI455X (gfx1250) — compile-verified, matrix operand order checked by reading, not run on hardware
//
#include <hip/hip_runtime.h>
#include <stdint.h>

// ---------------------------------------------------------------------------
// MHA forward for MI455X (gfx1250, wave32, WMMA, async global->LDS staging).
//   B=2, T=2048, D_MODEL=1024, H=16, d_k=64.
// f32->f16 convert -> WMMA GEMMs (Q/K head-major, V d-major) -> RoPE (score
// scale folded into Q) -> flash attention (WMMA, online softmax with DPP16
// row reductions, O' = Vt*Pt) -> WMMA GEMM output projection (fp32 out).
// Tile staging: global_load_async_to_lds_b128 (ASYNCcnt), triple buffered.
// ---------------------------------------------------------------------------

typedef __attribute__((ext_vector_type(16))) _Float16 v16h;
typedef __attribute__((ext_vector_type(8)))  float    v8f;

#define D_MODEL   1024
#define NUM_HEADS 16
#define D_K       64
#define BATCH     2
#define SEQ       2048
#define ROWS      (BATCH * SEQ)   // 4096

union Frag16 { uint32_t u[8]; uint4 q[2]; v16h v; };
union Pack8h { _Float16 h[8]; uint4 q; };

__device__ __forceinline__ int laneId() { return (int)(threadIdx.x & 31); }

__device__ __forceinline__ v8f wmma_f16(v16h a, v16h b, v8f c) {
  // D = A(16x32 f16) * B(32x16 f16) + C(16x16 f32)
  return __builtin_amdgcn_wmma_f32_16x16x32_f16(
      false, a, false, b, (short)0, c, false, false);
}

// 16-bit fragment from a row-major LDS tile [row][K] (strideH halves/row).
// ISA 7.12.2: lane L row = L&15; kHi = (L>=16)?8:0; VGPRs j<4 hold K kHi+0..7,
// j>=4 hold K 16+kHi+0..7 -> two 16B-contiguous groups => 2x ds_load_b128.
__device__ __forceinline__ v16h frag_rowmajor(const _Float16* sm, int row,
                                              int strideH, int offH) {
  int kHi = (laneId() >> 4) * 8;
  int base = row * strideH + offH + kHi;   // multiple of 8 halves (16B)
  Frag16 f;
  f.q[0] = *(const uint4*)(sm + base);
  f.q[1] = *(const uint4*)(sm + base + 16);
  return f.v;
}

// Async global->LDS copy of `bytes` contiguous bytes (16B granular).
// Uniform per-thread trip count => fixed ASYNCcnt increment per wave.
__device__ __forceinline__ void async_copy(const _Float16* g, _Float16* sm,
                                           int bytes, int tid, int nThr) {
  uint64_t gb = (uint64_t)(uintptr_t)g;
  uint32_t lb = (uint32_t)(uintptr_t)sm;   // low 32 bits = LDS offset
  for (int off = tid * 16; off < bytes; off += nThr * 16) {
    uint32_t lds = lb + (uint32_t)off;
    uint32_t voff = (uint32_t)off;
    asm volatile("global_load_async_to_lds_b128 %0, %1, %2"
                 :: "v"(lds), "v"(voff), "s"(gb) : "memory");
  }
}

// Async stage of an nRows x 32-half tile, global row stride gStrideH halves.
__device__ __forceinline__ void async_stage32(const _Float16* g, int gStrideH,
                                              _Float16* sm, int nRows,
                                              int tid, int nThr) {
  uint64_t gb = (uint64_t)(uintptr_t)g;
  uint32_t lb = (uint32_t)(uintptr_t)sm;
  int nCh = nRows * 4;                      // 16B chunks
  for (int c = tid; c < nCh; c += nThr) {
    int r = c >> 2, cc = c & 3;
    uint32_t voff = (uint32_t)(r * gStrideH * 2 + cc * 16);
    uint32_t lds = lb + (uint32_t)(c * 16);
    asm volatile("global_load_async_to_lds_b128 %0, %1, %2"
                 :: "v"(lds), "v"(voff), "s"(gb) : "memory");
  }
}

// --- 16-lane butterfly reductions via DPP16 (no DS traffic, no addr math) --
// xor1 = quad_perm(1,0,3,2)=0xB1, xor2 = quad_perm(2,3,0,1)=0x4E,
// 8-group = row_half_mirror (0x141), 16-group = row_mirror (0x140).
// All controls stay inside a 16-lane row => the two wave halves reduce
// independently, which is exactly what the score C-layout needs.
template <int CTRL>
__device__ __forceinline__ float dpp_mov(float v) {
  union { float f; int i; } a, b;
  a.f = v;
  b.i = __builtin_amdgcn_update_dpp(a.i, a.i, CTRL, 0xF, 0xF, true);
  return b.f;
}
__device__ __forceinline__ float redmax16(float v) {
  v = fmaxf(v, dpp_mov<0xB1>(v));
  v = fmaxf(v, dpp_mov<0x4E>(v));
  v = fmaxf(v, dpp_mov<0x141>(v));
  v = fmaxf(v, dpp_mov<0x140>(v));
  return v;
}
__device__ __forceinline__ float redsum16(float v) {
  v += dpp_mov<0xB1>(v);
  v += dpp_mov<0x4E>(v);
  v += dpp_mov<0x141>(v);
  v += dpp_mov<0x140>(v);
  return v;
}

// --------------------------------------------------------------------------
__global__ void f32_to_f16(const float* __restrict__ in,
                           _Float16* __restrict__ out, int n) {
  int i = blockIdx.x * blockDim.x + threadIdx.x;
  if (i < n) out[i] = (_Float16)in[i];
}

// In-place RoPE on [B*H, T, d_k] f16; `scale` folds 1/sqrt(d_k) into Q.
__global__ void rope_kernel(_Float16* __restrict__ buf, float scale) {
  int idx = blockIdx.x * blockDim.x + threadIdx.x;
  const int total = BATCH * NUM_HEADS * SEQ * (D_K / 2);
  if (idx >= total) return;
  int i = idx & 31;
  int t = (idx >> 5) & (SEQ - 1);
  int bh = idx >> 16;
  float ang = (float)t * __powf(10000.f, -((float)(2 * i)) / (float)D_K);
  float s, c;
  __sincosf(ang, &s, &c);
  int base = (bh * SEQ + t) * D_K + i;
  float x1 = (float)buf[base], x2 = (float)buf[base + 32];
  buf[base]      = (_Float16)((x1 * c - x2 * s) * scale);
  buf[base + 32] = (_Float16)((x1 * s + x2 * c) * scale);
}

// GEMM: C = A(4096x1024 f16) * W^T (W row-major 1024x1024 f16).
// Block = 4 waves = 64(M) x 64(N) tile; K-chunks of 32; triple-buffered
// async staging (A tile + B tile shared by all waves; 16 WMMAs / 8KB staged).
// MODE 0: f16 head-major [B,H,T,dk]; MODE 2: f16 d-major [B,H,dk,T];
// MODE 1: f32 flat [rows, D_MODEL].
template <int MODE>
__global__ __launch_bounds__(128) void gemm_wmma(
    const _Float16* __restrict__ A, const _Float16* __restrict__ W,
    void* __restrict__ outP) {
  __shared__ __align__(16) _Float16 smA[3][64 * 32];
  __shared__ __align__(16) _Float16 smB[3][64 * 32];
  int tid = threadIdx.x, wave = tid >> 5, lane = tid & 31, m = lane & 15;
  int rowBase = (int)(blockIdx.x >> 4) * 64;   // 64 row tiles
  int nBase   = (int)(blockIdx.x & 15) * 64;   // 16 col groups
  const int NK = D_MODEL / 32;

  // Prologue: batches for chunks 0 and 1 (4 async instr per wave per batch).
  async_stage32(A + rowBase * D_MODEL + 0,  D_MODEL, smA[0], 64, tid, 128);
  async_stage32(W + nBase  * D_MODEL + 0,  D_MODEL, smB[0], 64, tid, 128);
  async_stage32(A + rowBase * D_MODEL + 32, D_MODEL, smA[1], 64, tid, 128);
  async_stage32(W + nBase  * D_MODEL + 32, D_MODEL, smB[1], 64, tid, 128);

  v8f zero = {};
  v8f acc[4] = {zero, zero, zero, zero};
  int buf = 0, nxt = 2;                        // rotating ring indices
  for (int ki = 0; ki < NK; ++ki) {
    asm volatile("s_wait_asynccnt 0x4" ::: "memory");  // chunk ki landed
    __syncthreads();
    v16h aF = frag_rowmajor(smA[buf], wave * 16 + m, 32, 0);
#pragma unroll
    for (int sub = 0; sub < 4; ++sub)
      acc[sub] =
          wmma_f16(aF, frag_rowmajor(smB[buf], sub * 16 + m, 32, 0), acc[sub]);
    if (ki + 2 < NK) {   // all waves passed barrier => ring slot `nxt` free
      int k = (ki + 2) * 32;
      async_stage32(A + rowBase * D_MODEL + k, D_MODEL, smA[nxt], 64, tid, 128);
      async_stage32(W + nBase  * D_MODEL + k, D_MODEL, smB[nxt], 64, tid, 128);
    }
    buf = (buf == 2) ? 0 : buf + 1;
    nxt = (nxt == 2) ? 0 : nxt + 1;
  }

#pragma unroll
  for (int sub = 0; sub < 4; ++sub) {
#pragma unroll
    for (int r = 0; r < 8; ++r) {
      int row = rowBase + wave * 16 + ((lane < 16) ? r : r + 8);
      int col = nBase + sub * 16 + m;
      float v = acc[sub][r];
      if (MODE == 0) {
        int b = row >> 11, t = row & (SEQ - 1);
        int h = col >> 6,  d = col & (D_K - 1);
        ((_Float16*)outP)[((size_t)(b * NUM_HEADS + h) * SEQ + t) * D_K + d] =
            (_Float16)v;
      } else if (MODE == 2) {
        int b = row >> 11, t = row & (SEQ - 1);
        int h = col >> 6,  d = col & (D_K - 1);
        ((_Float16*)outP)[((size_t)(b * NUM_HEADS + h) * D_K + d) * SEQ + t] =
            (_Float16)v;
      } else {
        ((float*)outP)[(size_t)row * D_MODEL + col] = v;
      }
    }
  }
}

// Flash attention. Block = 4 waves = 64 query rows of one (b,h).
// Scores: S = Q*K^T (C-layout rows=q, cols=key). O computed transposed:
// O'(d,q) = Vt(d,key) x Pt(key,q); Vt tile is row-major d-major V, Pt operand
// is the sP [q][key] tile read with the row-major fragment gather.
__global__ __launch_bounds__(128) void flash_attn(
    const _Float16* __restrict__ Q, const _Float16* __restrict__ K,
    const _Float16* __restrict__ Vd, _Float16* __restrict__ O16) {
  __shared__ __align__(16) _Float16 sQ[64 * D_K];       // 8 KB
  __shared__ __align__(16) _Float16 sK[3][32 * D_K];    // 12 KB
  __shared__ __align__(16) _Float16 sV[3][64 * 32];     // 12 KB (V^T tiles)
  __shared__ __align__(16) _Float16 sP[4][16 * 32];     // 4 KB per-wave P
  __shared__ float sStat[4][16];
  int tid = threadIdx.x, wave = tid >> 5, lane = tid & 31, m = lane & 15;
  int bh = blockIdx.y, qBase = (int)blockIdx.x * 64;
  int b = bh >> 4, h = bh & 15;

  const _Float16* Kbh = K  + (size_t)bh * SEQ * D_K;    // [key][d]
  const _Float16* Vbh = Vd + (size_t)bh * D_K * SEQ;    // [d][t]

  // Prologue: Q block (4 instr/wave) + key-chunks 0,1 (4 instr/wave each).
  async_copy(Q + ((size_t)bh * SEQ + qBase) * D_K, sQ, 64 * D_K * 2, tid, 128);
  async_copy(Kbh, sK[0], 32 * D_K * 2, tid, 128);
  async_stage32(Vbh, SEQ, sV[0], 64, tid, 128);
  async_copy(Kbh + 32 * D_K, sK[1], 32 * D_K * 2, tid, 128);
  async_stage32(Vbh + 32, SEQ, sV[1], 64, tid, 128);
  asm volatile("s_wait_asynccnt 0x8" ::: "memory");      // Q block done
  __syncthreads();
  v16h qf0 = frag_rowmajor(sQ + wave * 16 * D_K, m, D_K, 0);
  v16h qf1 = frag_rowmajor(sQ + wave * 16 * D_K, m, D_K, 32);

  v8f zero = {};
  v8f accO[4] = {zero, zero, zero, zero};   // O' tiles: rows=d, cols=q
  float rm[8], rl[8];
#pragma unroll
  for (int r = 0; r < 8; ++r) { rm[r] = -1e30f; rl[r] = 0.f; }

  const int nkc = SEQ / 32;
  int buf = 0, nxt = 2;                      // rotating ring indices
  for (int kc = 0; kc < nkc; ++kc) {
    asm volatile("s_wait_asynccnt 0x4" ::: "memory");    // chunk kc landed
    __syncthreads();

    v8f s0 = zero, s1 = zero;                            // keys 0-15 / 16-31
    s0 = wmma_f16(qf0, frag_rowmajor(sK[buf], m, D_K, 0),  s0);
    s0 = wmma_f16(qf1, frag_rowmajor(sK[buf], m, D_K, 32), s0);
    s1 = wmma_f16(qf0, frag_rowmajor(sK[buf] + 16 * D_K, m, D_K, 0),  s1);
    s1 = wmma_f16(qf1, frag_rowmajor(sK[buf] + 16 * D_K, m, D_K, 32), s1);

    // Online softmax over this 32-key chunk (DPP16 row reductions).
#pragma unroll
    for (int r = 0; r < 8; ++r) {
      float tmax = redmax16(fmaxf(s0[r], s1[r]));
      float mn = fmaxf(rm[r], tmax);
      float corr = __expf(rm[r] - mn);
      float p0 = __expf(s0[r] - mn);
      float p1 = __expf(s1[r] - mn);
      rl[r] = rl[r] * corr + redsum16(p0 + p1);
      rm[r] = mn;
      int rr = (lane < 16) ? r : r + 8;
      sP[wave][rr * 32 + m]      = (_Float16)p0;
      sP[wave][rr * 32 + m + 16] = (_Float16)p1;
      if (m == r) sStat[wave][rr] = corr;                // 16 distinct writers
    }
    __syncthreads();

    float myCorr = sStat[wave][m];                        // corr for q = m
    v16h pfB = frag_rowmajor(sP[wave], m, 32, 0);         // Pt operand
#pragma unroll
    for (int g = 0; g < 4; ++g) {
#pragma unroll
      for (int r = 0; r < 8; ++r) accO[g][r] *= myCorr;
      accO[g] = wmma_f16(frag_rowmajor(sV[buf] + g * 16 * 32, m, 32, 0), pfB,
                         accO[g]);
    }

    if (kc + 2 < nkc) {   // ring slot `nxt` free: all waves past barrier
      async_copy(Kbh + (size_t)(kc + 2) * 32 * D_K, sK[nxt], 32 * D_K * 2,
                 tid, 128);
      async_stage32(Vbh + (kc + 2) * 32, SEQ, sV[nxt], 64, tid, 128);
    }
    buf = (buf == 2) ? 0 : buf + 1;
    nxt = (nxt == 2) ? 0 : nxt + 1;
  }

  // 1/l per q via the stat tile, then contiguous b128 stores of O rows.
#pragma unroll
  for (int r = 0; r < 8; ++r) {
    int rr = (lane < 16) ? r : r + 8;
    if (m == r) sStat[wave][rr] = 1.0f / rl[r];
  }
  __syncthreads();
  float inv = sStat[wave][m];
  int gRow = b * SEQ + qBase + wave * 16 + m;             // q = m
  int dHalf = (lane < 16) ? 0 : 8;
#pragma unroll
  for (int g = 0; g < 4; ++g) {
    Pack8h pk;
#pragma unroll
    for (int r = 0; r < 8; ++r) pk.h[r] = (_Float16)(accO[g][r] * inv);
    *(uint4*)(O16 + (size_t)gRow * D_MODEL + h * D_K + g * 16 + dHalf) = pk.q;
  }
}

// --------------------------------------------------------------------------
extern "C" void kernel_launch(void* const* d_in, const int* in_sizes, int n_in,
                              void* d_out, int out_size, void* d_ws,
                              size_t ws_size, hipStream_t stream) {
  (void)in_sizes; (void)n_in; (void)out_size; (void)ws_size;
  const float* x  = (const float*)d_in[0];
  const float* wq = (const float*)d_in[1];
  const float* wk = (const float*)d_in[2];
  const float* wv = (const float*)d_in[3];
  const float* wo = (const float*)d_in[4];

  char* ws = (char*)d_ws;
  size_t off = 0;
  auto alloc = [&](size_t bytes) {
    char* p = ws + off;
    off += (bytes + 255) & ~(size_t)255;
    return p;
  };
  _Float16* x16    = (_Float16*)alloc((size_t)ROWS * D_MODEL * 2);
  _Float16* wq16   = (_Float16*)alloc((size_t)D_MODEL * D_MODEL * 2);
  _Float16* wk16   = (_Float16*)alloc((size_t)D_MODEL * D_MODEL * 2);
  _Float16* wv16   = (_Float16*)alloc((size_t)D_MODEL * D_MODEL * 2);
  _Float16* wo16   = (_Float16*)alloc((size_t)D_MODEL * D_MODEL * 2);
  _Float16* Qb     = (_Float16*)alloc((size_t)ROWS * D_MODEL * 2); // [B,H,T,dk]
  _Float16* Kb     = (_Float16*)alloc((size_t)ROWS * D_MODEL * 2); // [B,H,T,dk]
  _Float16* Vb     = (_Float16*)alloc((size_t)ROWS * D_MODEL * 2); // [B,H,dk,T]
  _Float16* attn16 = (_Float16*)alloc((size_t)ROWS * D_MODEL * 2); // [B,T,D]

  const int nx = ROWS * D_MODEL, nw = D_MODEL * D_MODEL;
  f32_to_f16<<<(nx + 255) / 256, 256, 0, stream>>>(x,  x16,  nx);
  f32_to_f16<<<(nw + 255) / 256, 256, 0, stream>>>(wq, wq16, nw);
  f32_to_f16<<<(nw + 255) / 256, 256, 0, stream>>>(wk, wk16, nw);
  f32_to_f16<<<(nw + 255) / 256, 256, 0, stream>>>(wv, wv16, nw);
  f32_to_f16<<<(nw + 255) / 256, 256, 0, stream>>>(wo, wo16, nw);

  gemm_wmma<0><<<1024, 128, 0, stream>>>(x16, wq16, Qb);
  gemm_wmma<0><<<1024, 128, 0, stream>>>(x16, wk16, Kb);
  gemm_wmma<2><<<1024, 128, 0, stream>>>(x16, wv16, Vb);

  const int nrope = BATCH * NUM_HEADS * SEQ * (D_K / 2);
  rope_kernel<<<(nrope + 255) / 256, 256, 0, stream>>>(Qb, 0.125f);
  rope_kernel<<<(nrope + 255) / 256, 256, 0, stream>>>(Kb, 1.0f);

  dim3 g(SEQ / 64, BATCH * NUM_HEADS);
  flash_attn<<<g, 128, 0, stream>>>(Qb, Kb, Vb, attn16);

  gemm_wmma<1><<<1024, 128, 0, stream>>>(attn16, wo16, d_out);
}
